// ModalSynth_910533066852
// MI455X (gfx1250) — compile-verified, matrix operand order checked by reading, not run on hardware
//
#include <hip/hip_runtime.h>
#include <hip/hip_bf16.h>

// ModalSynth for MI455X (gfx1250, wave32).
//
// Key idea: the reference's 48000-long cumsum per (batch,mode) collapses to a
// 400-long knot prefix because interp scale is exactly 1/120:
//   j < 60           : w = f[0]                      (head)
//   j in segment i   : w = f[i] + (f[i+1]-f[i])*(r+0.5)/120,  r = (j-60)%120
//   j >= 47940       : w = f[399]                    (tail)
// cumsum(w)[j] (inclusive):
//   head : (j+1)*f0
//   seg  : PF[i] + q*f[i] + (f[i+1]-f[i])*q*q/240,   q = r+1
//   tail : PF[399] + (j-47939)*f[399]
// with PF[i] = 60*f0 + 60*sum_{k<i}(f[k]+f[k+1])  (PF[0] = 60*f0).
//
// Phase 1 builds PF (1024 x 400 floats = 1.6 MB in d_ws).
// Phase 2: block = 256 thr (8 waves) handles (batch b, 128-sample tile).
//   Lane owns sample t = tile + wave*16 + (lane&15) and 32 of 64 modes.
//   Mode reduction runs on the matrix pipe: two v_wmma_f32_16x16x32_f16 with
//   an all-ones A; B column N = lane%16 = sample, K slots = modes; the f32
//   accumulator's c[0] on every lane is the 64-mode sum for its column.

typedef __attribute__((ext_vector_type(16))) _Float16 v16h;
typedef __attribute__((ext_vector_type(8)))  float    v8f;

#define NS     48000
#define NKNOT  400
#define NMODE  64
#define NBATCH 16
#define TILE   128   // samples per block (8 waves * 16)

// ---------------- Phase 1: knot-level prefix of the frequency track --------
__global__ __launch_bounds__(256) void ModalSynth_knot_prefix(
    const float* __restrict__ params, float* __restrict__ PF)
{
    int idx = blockIdx.x * blockDim.x + threadIdx.x;   // 0..1023 = (b*64+m)
    if (idx >= NBATCH * NMODE) return;
    int b = idx >> 6, m = idx & 63;
    const float* F = params + ((size_t)(b * 3 + 0) * NMODE + m) * NKNOT;
    float* row = PF + (size_t)idx * NKNOT;

    float prev = F[0];
    float acc  = 60.0f * prev;        // PF[0] = 60*f0 (head contribution)
    row[0] = acc;
    for (int i = 1; i < NKNOT; ++i) {
        float fi = F[i];
        acc += 60.0f * (prev + fi);   // segment (i-1) sum = 60*(f[i-1]+f[i])
        row[i] = acc;
        prev = fi;
    }
}

// ---------------- Phase 2: evaluate + WMMA mode reduction ------------------
__global__ __launch_bounds__(256) void ModalSynth_eval(
    const float* __restrict__ params, const float* __restrict__ PF,
    float* __restrict__ out)
{
    const int b    = blockIdx.y;
    const int j0   = blockIdx.x * TILE;
    const int lane = threadIdx.x & 31;
    const int wave = threadIdx.x >> 5;

    // A 128-sample window can touch up to 3 segments -> 4 knots, 3 PF entries.
    __shared__ float sF [4 * NMODE];
    __shared__ float sA [4 * NMODE];
    __shared__ float sPF[3 * NMODE];
    __shared__ float sPh[NMODE];

    int iMin = (j0 - 60) / 120;
    if (iMin < 0)   iMin = 0;
    if (iMin > 398) iMin = 398;

    if (threadIdx.x < NMODE) {
        int m = threadIdx.x;
        const float* Frow  = params + ((size_t)(b * 3 + 0) * NMODE + m) * NKNOT;
        const float* Arow  = params + ((size_t)(b * 3 + 1) * NMODE + m) * NKNOT;
        const float* Prow  = params + ((size_t)(b * 3 + 2) * NMODE + m) * NKNOT;
        const float* PFrow = PF + (size_t)(b * NMODE + m) * NKNOT;
#pragma unroll
        for (int lk = 0; lk < 4; ++lk) {
            int g = iMin + lk; if (g > 399) g = 399;
            sF[lk * NMODE + m] = Frow[g];
            sA[lk * NMODE + m] = Arow[g];
        }
#pragma unroll
        for (int lk = 0; lk < 3; ++lk) {
            int g = iMin + lk; if (g > 399) g = 399;
            sPF[lk * NMODE + m] = PFrow[g];
        }
        sPh[m] = Prow[0];
    }
    __syncthreads();

    // Per-lane sample + region decode (branch-free mode loop afterwards).
    const int j = j0 + wave * 16 + (lane & 15);
    int l0, l1, lpf;                  // local knot / PF indices into LDS
    float q, c2, afrac;
    if (j < 60) {                                        // head: w=f0, a=g0
        l0 = 0 - iMin; l1 = l0; lpf = -1;
        q = (float)(j + 1); c2 = 0.0f; afrac = 0.0f;
    } else if (j >= NS - 60) {                           // tail: w=f399
        l0 = 399 - iMin; l1 = l0; lpf = 399 - iMin;
        q = (float)(j - (NS - 61)); c2 = 0.0f; afrac = 0.0f;
    } else {                                             // interior segment
        int i = (j - 60) / 120;
        int r = (j - 60) - i * 120;
        l0 = i - iMin; l1 = l0 + 1; lpf = l0;
        float qq = (float)(r + 1);
        q = qq; c2 = qq * qq * (1.0f / 240.0f);
        afrac = ((float)r + 0.5f) * (1.0f / 120.0f);
    }

    // Modes: lanes {L, L+16} share sample column L%16; lane group picks 32 modes.
    const int mbase = (lane >> 4) * 32;
    v16h s0 = {}, s1 = {};
#pragma unroll
    for (int k = 0; k < 32; ++k) {
        int m = mbase + k;
        float fi = sF[l0 * NMODE + m];
        float fj = sF[l1 * NMODE + m];
        float ai = sA[l0 * NMODE + m];
        float aj = sA[l1 * NMODE + m];
        float base = (lpf >= 0) ? sPF[lpf * NMODE + m] : 0.0f;
        float ph  = sPh[m] + base + q * fi + c2 * (fj - fi);
        float amp = ai + afrac * (aj - ai);
        float s   = amp * __sinf(ph);
        ((k < 16) ? s0 : s1)[k & 15] = (_Float16)s;
    }

    // Ones-matmul reduction over K (= modes): D[i,t] = sum_K B[K,t] for all i.
    v16h ones;
#pragma unroll
    for (int k = 0; k < 16; ++k) ones[k] = (_Float16)1.0f;

    v8f c = {};
    c = __builtin_amdgcn_wmma_f32_16x16x32_f16(false, ones, false, s0,
                                               (short)0, c, false, false);
    c = __builtin_amdgcn_wmma_f32_16x16x32_f16(false, ones, false, s1,
                                               (short)0, c, false, false);

    // c[0] on lane L holds the column (L%16) sum; lanes 16..31 are duplicates.
    if (lane < 16) out[(size_t)b * NS + j] = c[0];
}

// ---------------------------------------------------------------------------
extern "C" void kernel_launch(void* const* d_in, const int* in_sizes, int n_in,
                              void* d_out, int out_size, void* d_ws, size_t ws_size,
                              hipStream_t stream) {
    (void)in_sizes; (void)n_in; (void)out_size; (void)ws_size;
    const float* params = (const float*)d_in[0];   // (16,3,64,400) f32
    float* out = (float*)d_out;                    // (16,1,48000)  f32
    float* PF  = (float*)d_ws;                     // 1024*400 f32 = 1.6 MB

    ModalSynth_knot_prefix<<<dim3((NBATCH * NMODE + 255) / 256), dim3(256), 0, stream>>>(
        params, PF);
    ModalSynth_eval<<<dim3(NS / TILE, NBATCH), dim3(256), 0, stream>>>(
        params, PF, out);
}
